// HSIC_26061861552238
// MI455X (gfx1250) — matrix operand order repaired and checked by reading, not compiled, not run on hardware
//
#include <hip/hip_runtime.h>

// HSIC, N=4096 D=1024, fp32 in/out.
// Fused: never materializes K matrices. Gram tiles via bf16 hi/lo split
// (3x v_wmma_f32_16x16x32_bf16 per fp32 GEMM step), exp + reductions fused
// into the tile epilogue. Centering reduced analytically to 5 global sums.

#define NN  4096
#define DD  1024
#define BM  128   // workgroup tile rows
#define BN  64    // workgroup tile cols
#define KB  32    // K per WMMA step (bf16 16x16x32)

typedef __attribute__((ext_vector_type(16))) __bf16 v16bf;
typedef __attribute__((ext_vector_type(8)))  __bf16 bf16x8;
typedef __attribute__((ext_vector_type(4)))  __bf16 bf16x4;
typedef __attribute__((ext_vector_type(8)))  float  v8f;

union FragU { bf16x8 h[2]; v16bf v; };

#define WMMA_BF16(A, B, C) \
  __builtin_amdgcn_wmma_f32_16x16x32_bf16(false, (A), false, (B), (short)0, (C), false, false)

// A-matrix 16x32 bf16 fragment (ISA 7.12.2): lane half h holds K runs
// [8h..8h+7] and [16+8h..16+8h+7] for row M = lane%16. Contiguous -> 2x b128.
__device__ __forceinline__ v16bf lds_frag_a(const __bf16* s, int row, int half) {
  const __bf16* p = s + row * KB;
  FragU f;
  f.h[0] = *(const bf16x8*)(p + half * 8);
  f.h[1] = *(const bf16x8*)(p + 16 + half * 8);
  return f.v;
}

// B-matrix 32x16 bf16 fragment: lane half h holds K runs [8h..8h+7] and
// [16+8h..16+8h+7] for column N = lane%16 (col panel stored row-major [col][K]). 2x b128.
__device__ __forceinline__ v16bf lds_frag_b(const __bf16* s, int col, int half) {
  const __bf16* p = s + col * KB;
  FragU f;
  f.h[0] = *(const bf16x8*)(p + half * 8);
  f.h[1] = *(const bf16x8*)(p + 16 + half * 8);
  return f.v;
}

// Stage nrows x 32 fp32 from global into LDS as bf16 hi + residual lo.
__device__ __forceinline__ void stage_tile(const float* __restrict__ g, int row0,
                                           int nrows, int kb,
                                           __bf16* shi, __bf16* slo, int tid) {
  const int nf4 = nrows * (KB / 4);
  for (int idx = tid; idx < nf4; idx += 256) {
    const int r  = idx >> 3;
    const int c4 = (idx & 7) << 2;
    const float4 v = *(const float4*)(g + (size_t)(row0 + r) * DD + kb + c4);
    bf16x4 h, l;
    h.x = (__bf16)v.x;  l.x = (__bf16)(v.x - (float)h.x);
    h.y = (__bf16)v.y;  l.y = (__bf16)(v.y - (float)h.y);
    h.z = (__bf16)v.z;  l.z = (__bf16)(v.z - (float)h.z);
    h.w = (__bf16)v.w;  l.w = (__bf16)(v.w - (float)h.w);
    *(bf16x4*)(shi + r * KB + c4) = h;
    *(bf16x4*)(slo + r * KB + c4) = l;
  }
}

__global__ void zero_kernel(float* __restrict__ p, int n) {
  int i = blockIdx.x * blockDim.x + threadIdx.x;
  if (i < n) p[i] = 0.0f;
}

// sq[i] = sum_k src[i,k]^2  (fp32). Blocks [0,NN) -> x, [NN,2NN) -> y.
__global__ __launch_bounds__(256)
void row_norms_kernel(const float* __restrict__ x, const float* __restrict__ y,
                      float* __restrict__ sqx, float* __restrict__ sqy) {
  const int b = blockIdx.x;
  const float* src = (b < NN) ? x : y;
  float* dst       = (b < NN) ? sqx : sqy;
  const int row    = (b < NN) ? b : (b - NN);
  float s = 0.0f;
  for (int k = threadIdx.x; k < DD; k += 256) {
    const float v = src[(size_t)row * DD + k];
    s += v * v;
  }
  for (int m = 1; m < 32; m <<= 1) s += __shfl_xor(s, m);
  __shared__ float red[8];
  if ((threadIdx.x & 31) == 0) red[threadIdx.x >> 5] = s;
  __syncthreads();
  if (threadIdx.x < 8) {
    float v = red[threadIdx.x];
    v += __shfl_xor(v, 1);
    v += __shfl_xor(v, 2);
    v += __shfl_xor(v, 4);
    if (threadIdx.x == 0) dst[row] = v;
  }
}

// Main fused tile kernel. Grid (NN/BN, NN/BM), 256 threads (8 wave32, 4x2).
__global__ __launch_bounds__(256)
void hsic_tile_kernel(const float* __restrict__ x, const float* __restrict__ y,
                      const float* __restrict__ sqx, const float* __restrict__ sqy,
                      float* __restrict__ rA,  // rowsum Kx  [m]
                      float* __restrict__ cA,  // colsum Kx  [n]
                      float* __restrict__ bM,  // rowsum KyT [m] = colsum Ky
                      float* __restrict__ bN,  // colsum KyT [n] = rowsum Ky
                      float* __restrict__ t1_out) {
  __shared__ __bf16 s_xr_h[BM * KB], s_xr_l[BM * KB];
  __shared__ __bf16 s_yr_h[BM * KB], s_yr_l[BM * KB];
  __shared__ __bf16 s_xc_h[BN * KB], s_xc_l[BN * KB];
  __shared__ __bf16 s_yc_h[BN * KB], s_yc_l[BN * KB];

  const int tid  = threadIdx.x;
  const int lane = tid & 31;
  const int wave = tid >> 5;
  const int half = lane >> 4;
  const int lm   = lane & 15;
  const int wm   = wave & 3;   // wave row 0..3 -> 32-row patch
  const int wn   = wave >> 2;  // wave col 0..1 -> 32-col patch
  const int tileM = blockIdx.y * BM;
  const int tileN = blockIdx.x * BN;

  v8f accX[2][2] = {};
  v8f accY[2][2] = {};

  for (int kb = 0; kb < DD; kb += KB) {
    __syncthreads();
    stage_tile(x, tileM, BM, kb, s_xr_h, s_xr_l, tid);
    stage_tile(y, tileM, BM, kb, s_yr_h, s_yr_l, tid);
    stage_tile(x, tileN, BN, kb, s_xc_h, s_xc_l, tid);
    stage_tile(y, tileN, BN, kb, s_yc_h, s_yc_l, tid);
    __syncthreads();

    v16bf axh[2], axl[2], ayh[2], ayl[2];
    v16bf bxh[2], bxl[2], byh[2], byl[2];
#pragma unroll
    for (int t = 0; t < 2; ++t) {
      const int ar = wm * 32 + t * 16 + lm;
      axh[t] = lds_frag_a(s_xr_h, ar, half);
      axl[t] = lds_frag_a(s_xr_l, ar, half);
      ayh[t] = lds_frag_a(s_yr_h, ar, half);
      ayl[t] = lds_frag_a(s_yr_l, ar, half);
      const int bc = wn * 32 + t * 16 + lm;
      bxh[t] = lds_frag_b(s_xc_h, bc, half);
      bxl[t] = lds_frag_b(s_xc_l, bc, half);
      byh[t] = lds_frag_b(s_yc_h, bc, half);
      byl[t] = lds_frag_b(s_yc_l, bc, half);
    }
#pragma unroll
    for (int ti = 0; ti < 2; ++ti) {
#pragma unroll
      for (int tj = 0; tj < 2; ++tj) {
        // fp32 ~= hi*hi + hi*lo + lo*hi (bf16x3 split)
        accX[ti][tj] = WMMA_BF16(axh[ti], bxh[tj], accX[ti][tj]);
        accX[ti][tj] = WMMA_BF16(axh[ti], bxl[tj], accX[ti][tj]);
        accX[ti][tj] = WMMA_BF16(axl[ti], bxh[tj], accX[ti][tj]);
        accY[ti][tj] = WMMA_BF16(ayh[ti], byh[tj], accY[ti][tj]);
        accY[ti][tj] = WMMA_BF16(ayh[ti], byl[tj], accY[ti][tj]);
        accY[ti][tj] = WMMA_BF16(ayl[ti], byh[tj], accY[ti][tj]);
      }
    }
  }

  // Epilogue. C/D layout: VGPR r, lanes 0-15 -> (M=r, N=lane),
  // lanes 16-31 -> (M=r+8, N=lane-16).
  float t1 = 0.0f;
#pragma unroll
  for (int ti = 0; ti < 2; ++ti) {
    const int mbase = tileM + wm * 32 + ti * 16 + half * 8;
    float sqm[8];
#pragma unroll
    for (int r = 0; r < 8; ++r) sqm[r] = sqx[mbase + r];
#pragma unroll
    for (int tj = 0; tj < 2; ++tj) {
      const int nidx = tileN + wn * 32 + tj * 16 + lm;
      const float sqn = sqy[nidx];
      float csA = 0.0f, csB = 0.0f;
#pragma unroll
      for (int r = 0; r < 8; ++r) {
        const float kx = __expf(accX[ti][tj][r] - sqm[r]);   // Kx[m, n]
        const float ky = __expf(accY[ti][tj][r] - sqn);      // Ky[n, m] = KyT[m, n]
        t1 += kx * ky;
        float rsx = kx, rsy = ky;  // reduce over the 16 lanes of this half (fixed m)
        rsx += __shfl_xor(rsx, 1); rsy += __shfl_xor(rsy, 1);
        rsx += __shfl_xor(rsx, 2); rsy += __shfl_xor(rsy, 2);
        rsx += __shfl_xor(rsx, 4); rsy += __shfl_xor(rsy, 4);
        rsx += __shfl_xor(rsx, 8); rsy += __shfl_xor(rsy, 8);
        if (lm == 0) {
          atomicAdd(&rA[mbase + r], rsx);
          atomicAdd(&bM[mbase + r], rsy);
        }
        csA += kx;
        csB += ky;
      }
      csA += __shfl_xor(csA, 16);  // combine halves (fixed n, all 16 m)
      csB += __shfl_xor(csB, 16);
      if (half == 0) {
        atomicAdd(&cA[nidx], csA);
        atomicAdd(&bN[nidx], csB);
      }
    }
  }
#pragma unroll
  for (int m = 1; m < 32; m <<= 1) t1 += __shfl_xor(t1, m);
  if (lane == 0) atomicAdd(t1_out, t1);
}

// S = T1 - (1/n)Σ cA·bN - (1/n)Σ rA·bM + (ΣbN / n^2)·(ΣrA);  out = S/(n-1)^2
__global__ __launch_bounds__(1024)
void finalize_kernel(const float* __restrict__ rA, const float* __restrict__ cA,
                     const float* __restrict__ bM, const float* __restrict__ bN,
                     const float* __restrict__ t1p, float* __restrict__ out) {
  float p1 = 0.0f, p2 = 0.0f, p3 = 0.0f, p4 = 0.0f;
  for (int i = threadIdx.x; i < NN; i += 1024) {
    p1 += cA[i] * bN[i];
    p2 += rA[i] * bM[i];
    p3 += rA[i];
    p4 += bN[i];
  }
  for (int m = 1; m < 32; m <<= 1) {
    p1 += __shfl_xor(p1, m);
    p2 += __shfl_xor(p2, m);
    p3 += __shfl_xor(p3, m);
    p4 += __shfl_xor(p4, m);
  }
  __shared__ float r1[32], r2[32], r3[32], r4[32];
  const int w = threadIdx.x >> 5, l = threadIdx.x & 31;
  if (l == 0) { r1[w] = p1; r2[w] = p2; r3[w] = p3; r4[w] = p4; }
  __syncthreads();
  if (threadIdx.x < 32) {
    p1 = r1[threadIdx.x]; p2 = r2[threadIdx.x];
    p3 = r3[threadIdx.x]; p4 = r4[threadIdx.x];
    for (int m = 1; m < 32; m <<= 1) {
      p1 += __shfl_xor(p1, m);
      p2 += __shfl_xor(p2, m);
      p3 += __shfl_xor(p3, m);
      p4 += __shfl_xor(p4, m);
    }
    if (threadIdx.x == 0) {
      const float n = (float)NN;
      const float S = *t1p - p1 / n - p2 / n + (p4 / (n * n)) * p3;
      out[0] = S / ((n - 1.0f) * (n - 1.0f));
    }
  }
}

extern "C" void kernel_launch(void* const* d_in, const int* in_sizes, int n_in,
                              void* d_out, int out_size, void* d_ws, size_t ws_size,
                              hipStream_t stream) {
  const float* x = (const float*)d_in[0];
  const float* y = (const float*)d_in[1];
  float* ws  = (float*)d_ws;
  float* sqx = ws;            // [NN]
  float* sqy = ws + NN;       // [NN]
  float* rA  = ws + 2 * NN;   // [NN]
  float* cA  = ws + 3 * NN;   // [NN]
  float* bM  = ws + 4 * NN;   // [NN]
  float* bN  = ws + 5 * NN;   // [NN]
  float* t1  = ws + 6 * NN;   // [1]

  const int nzero = 4 * NN + 1;  // rA, cA, bM, bN, t1 (contiguous)
  zero_kernel<<<(nzero + 255) / 256, 256, 0, stream>>>(rA, nzero);
  row_norms_kernel<<<2 * NN, 256, 0, stream>>>(x, y, sqx, sqy);
  dim3 grid(NN / BN, NN / BM);
  hsic_tile_kernel<<<grid, 256, 0, stream>>>(x, y, sqx, sqy, rA, cA, bM, bN, t1);
  finalize_kernel<<<1, 1024, 0, stream>>>(rA, cA, bM, bN, t1, (float*)d_out);
}
